// SparseAutoEncoder_84920093376851
// MI455X (gfx1250) — matrix operand order repaired and checked by reading, not compiled, not run on hardware
//
#include <hip/hip_runtime.h>
#include <hip/hip_bf16.h>
#include <math.h>

#define IN_DIM  4096
#define LAT_DIM 40960
#define BATCH   256
#define TOPK    16384
#define N_TOT   (BATCH * LAT_DIM)
#define KCHUNKS 8
#define KCH     (LAT_DIM / KCHUNKS)   // 5120

typedef __attribute__((ext_vector_type(16))) __bf16 v16bf;
typedef __attribute__((ext_vector_type(8)))  float  v8f;
typedef __attribute__((ext_vector_type(4)))  float  f32x4;
typedef __attribute__((ext_vector_type(4)))  __bf16 bf16x4;

union V16 { v16bf v; bf16x4 q[4]; };

__device__ __forceinline__ unsigned mono_key(float f) {
  unsigned u = __float_as_uint(f);
  return (u & 0x80000000u) ? ~u : (u | 0x80000000u);
}

// A fragment (16-bit A 16x32 layout): lane holds row m=lane&15,
// contiguous K-run [16*kg,+16). p points at (row base + k0).
__device__ __forceinline__ v16bf load_a_frag(const __bf16* __restrict__ p, int kg) {
  V16 a;
  a.q[0] = *(const bf16x4*)(p + 16 * kg);
  a.q[1] = *(const bf16x4*)(p + 16 * kg + 4);
  a.q[2] = *(const bf16x4*)(p + 16 * kg + 8);
  a.q[3] = *(const bf16x4*)(p + 16 * kg + 12);
  return a.v;
}

// Raw f32 B stage (16-bit B 32x16 layout): lane holds col n=lane&15,
// contiguous K-run [16*kg,+16). Converted to bf16 only at use, so the
// pipelined next-stage load's wait sinks behind the current WMMAs.
struct BStage { f32x4 r[4]; };

__device__ __forceinline__ BStage load_b_raw(const float* __restrict__ p, int kg) {
  const f32x4* br = (const f32x4*)(p + 16 * kg);
  BStage s;
  s.r[0] = br[0]; s.r[1] = br[1]; s.r[2] = br[2]; s.r[3] = br[3];
  return s;
}

__device__ __forceinline__ v16bf cvt_b(const BStage& s) {
  V16 b;
  b.q[0] = __builtin_convertvector(s.r[0], bf16x4);
  b.q[1] = __builtin_convertvector(s.r[1], bf16x4);
  b.q[2] = __builtin_convertvector(s.r[2], bf16x4);
  b.q[3] = __builtin_convertvector(s.r[3], bf16x4);
  return b.v;
}

// ---------------------------------------------------------------------------
// x (f32, 256x4096) -> bf16 once; removes all A-side VALU from the encoder.
// ---------------------------------------------------------------------------
__global__ __launch_bounds__(256) void cvt_x(const float* __restrict__ X,
                                             __bf16* __restrict__ XB) {
  const int n4 = (BATCH * IN_DIM) / 4;
  for (int idx = blockIdx.x * blockDim.x + threadIdx.x; idx < n4;
       idx += gridDim.x * blockDim.x)
    ((bf16x4*)XB)[idx] = __builtin_convertvector(((const f32x4*)X)[idx], bf16x4);
}

// ---------------------------------------------------------------------------
// Ping-pong double-buffered 64x16 MMA strip: 4 M-tiles per wave, B fragment
// reused 4x. The 2x-unrolled loop alternates buffer roles by NAME, so no
// register-rotation movs are needed. KLEN must be a multiple of 64.
// ---------------------------------------------------------------------------
template <int KLEN>
__device__ __forceinline__ void mma_strip(const __bf16* __restrict__ a0,
                                          size_t tstride,  // A tile stride (elems)
                                          const float* __restrict__ bp,
                                          int kg, v8f acc[4]) {
  static_assert(KLEN % 64 == 0, "KLEN must be a multiple of 64");
  BStage b0 = load_b_raw(bp, kg);
  v16bf f0[4], f1[4];
  BStage b1;
#pragma unroll
  for (int t = 0; t < 4; ++t) f0[t] = load_a_frag(a0 + tstride * t, kg);

  for (int k0 = 0; k0 < KLEN - 64; k0 += 64) {
    __builtin_prefetch(bp + k0 + 512, 0, 0);   // global_prefetch_b8, speculative
    // Stage 0: load (k0+32) into buf1, compute buf0.
    b1 = load_b_raw(bp + k0 + 32, kg);
#pragma unroll
    for (int t = 0; t < 4; ++t) f1[t] = load_a_frag(a0 + tstride * t + k0 + 32, kg);
    {
      const v16bf b = cvt_b(b0);
#pragma unroll
      for (int t = 0; t < 4; ++t)
        acc[t] = __builtin_amdgcn_wmma_f32_16x16x32_bf16(
            false, f0[t], false, b, (short)0, acc[t], false, false);
    }
    // Stage 1: load (k0+64) into buf0, compute buf1.
    b0 = load_b_raw(bp + k0 + 64, kg);
#pragma unroll
    for (int t = 0; t < 4; ++t) f0[t] = load_a_frag(a0 + tstride * t + k0 + 64, kg);
    {
      const v16bf b = cvt_b(b1);
#pragma unroll
      for (int t = 0; t < 4; ++t)
        acc[t] = __builtin_amdgcn_wmma_f32_16x16x32_bf16(
            false, f1[t], false, b, (short)0, acc[t], false, false);
    }
  }
  // Tail (k0 = KLEN-64): no shadow load past the end.
  b1 = load_b_raw(bp + KLEN - 32, kg);
#pragma unroll
  for (int t = 0; t < 4; ++t) f1[t] = load_a_frag(a0 + tstride * t + KLEN - 32, kg);
  {
    const v16bf b = cvt_b(b0);
#pragma unroll
    for (int t = 0; t < 4; ++t)
      acc[t] = __builtin_amdgcn_wmma_f32_16x16x32_bf16(
          false, f0[t], false, b, (short)0, acc[t], false, false);
  }
  {
    const v16bf b = cvt_b(b1);
#pragma unroll
    for (int t = 0; t < 4; ++t)
      acc[t] = __builtin_amdgcn_wmma_f32_16x16x32_bf16(
          false, f1[t], false, b, (short)0, acc[t], false, false);
  }
}

// ---------------------------------------------------------------------------
// Encoder GEMM: pre[m][n] = sum_k x[m][k] * W_enc[n][k] + b_enc[n]
// ---------------------------------------------------------------------------
__global__ __launch_bounds__(256) void enc_gemm(const __bf16* __restrict__ XB,
                                                const float* __restrict__ W,
                                                const float* __restrict__ Bv,
                                                float* __restrict__ OUT) {
  const int wave = blockIdx.x * (blockDim.x >> 5) + (threadIdx.x >> 5);
  const int lane = threadIdx.x & 31;
  const int row  = lane & 15;
  const int kg   = lane >> 4;
  const int m0 = (wave & 3) * 64;   // 4 m-tile groups
  const int n0 = (wave >> 2) * 16;  // 2560 n-tiles

  const __bf16* a0 = XB + (size_t)(m0 + row) * IN_DIM;
  const float*  bp = W + (size_t)(n0 + row) * IN_DIM;

  v8f acc[4] = {};
  mma_strip<IN_DIM>(a0, (size_t)16 * IN_DIM, bp, kg, acc);

  const int n = n0 + row;
  const float bias = Bv[n];
#pragma unroll
  for (int t = 0; t < 4; ++t)
#pragma unroll
    for (int r = 0; r < 8; ++r)
      OUT[(size_t)(m0 + t * 16 + kg * 8 + r) * LAT_DIM + n] = acc[t][r] + bias;
}

// ---------------------------------------------------------------------------
// Radix top-k select (3 rounds over monotone u32 keys: 11 + 11 + 10 bits)
// sel[0]=prefix key, sel[1]=k remaining, sel[2]=final threshold key,
// sel[3]=tie counter (atomic), sel[4]=ties allowed
// ---------------------------------------------------------------------------
__global__ void topk_init(unsigned* __restrict__ hist, unsigned* __restrict__ sel) {
  for (int i = threadIdx.x; i < 2048; i += blockDim.x) hist[i] = 0;
  if (threadIdx.x == 0) { sel[0] = 0u; sel[1] = TOPK; sel[3] = 0u; }
}

__global__ __launch_bounds__(256) void topk_hist(const float* __restrict__ pre,
                                                 unsigned* __restrict__ hist,
                                                 const unsigned* __restrict__ sel,
                                                 int shift, int bits) {
  __shared__ unsigned lh[2048];
  for (int i = threadIdx.x; i < 2048; i += blockDim.x) lh[i] = 0;
  __syncthreads();
  const unsigned prefix = sel[0];
  const int fieldTop = shift + bits;
  const unsigned pmask = (fieldTop >= 32) ? 0u : ~((1u << fieldTop) - 1u);
  const unsigned bmask = (1u << bits) - 1u;
  for (int i = blockIdx.x * blockDim.x + threadIdx.x; i < N_TOT;
       i += gridDim.x * blockDim.x) {
    const unsigned key = mono_key(pre[i]);
    if ((key & pmask) == prefix)
      atomicAdd(&lh[(key >> shift) & bmask], 1u);
  }
  __syncthreads();
  for (int i = threadIdx.x; i < 2048; i += blockDim.x)
    if (lh[i]) atomicAdd(&hist[i], lh[i]);
}

__global__ void topk_scan(unsigned* __restrict__ hist, unsigned* __restrict__ sel,
                          int shift, int bits, int lastRound) {
  if (threadIdx.x != 0 || blockIdx.x != 0) return;
  const unsigned kRem = sel[1];
  unsigned cum = 0;
  int binSel = 0;
  for (int b = (1 << bits) - 1; b >= 0; --b) {
    const unsigned c = hist[b];
    if (cum + c >= kRem) { binSel = b; break; }
    cum += c;
  }
  sel[0] |= ((unsigned)binSel << shift);
  sel[1] = kRem - cum;
  if (lastRound) { sel[2] = sel[0]; sel[4] = sel[1]; sel[3] = 0u; }
  for (int i = 0; i < 2048; ++i) hist[i] = 0;  // reset for next round
}

// Keep strictly-above-threshold values plus a tie budget at the threshold;
// emit the sparse latent directly in bf16 for the decoder GEMM.
__global__ __launch_bounds__(256) void topk_scatter(const float* __restrict__ pre,
                                                    __bf16* __restrict__ latb,
                                                    unsigned* __restrict__ sel) {
  const unsigned T = sel[2];
  const unsigned allowed = sel[4];
  for (int i = blockIdx.x * blockDim.x + threadIdx.x; i < N_TOT;
       i += gridDim.x * blockDim.x) {
    const float v = pre[i];
    const unsigned key = mono_key(v);
    bool keep = key > T;
    if (key == T) keep = (atomicAdd(&sel[3], 1u) < allowed);
    latb[i] = keep ? (__bf16)v : (__bf16)0.0f;
  }
}

// ---------------------------------------------------------------------------
// Decoder GEMM (K-split): partial[kc][m][n] = sum_{k in chunk} lat[m][k]*W_dec[n][k]
// 4 m-tiles x 8 K-chunks x 256 n-tiles = 8192 waves; fixed-order reduction
// in dec_reduce keeps the result bitwise deterministic (no float atomics).
// ---------------------------------------------------------------------------
__global__ __launch_bounds__(256) void dec_gemm(const __bf16* __restrict__ L,
                                                const float* __restrict__ W,
                                                float* __restrict__ PART) {
  const int wave = blockIdx.x * (blockDim.x >> 5) + (threadIdx.x >> 5);
  const int lane = threadIdx.x & 31;
  const int row  = lane & 15;
  const int kg   = lane >> 4;
  const int m0 = (wave & 3) * 64;
  const int kc = (wave >> 2) & (KCHUNKS - 1);
  const int n0 = (wave >> 5) * 16;  // 256 n-tiles
  const int kbase = kc * KCH;

  const __bf16* a0 = L + (size_t)(m0 + row) * LAT_DIM + kbase;
  const float*  bp = W + (size_t)(n0 + row) * LAT_DIM + kbase;

  v8f acc[4] = {};
  mma_strip<KCH>(a0, (size_t)16 * LAT_DIM, bp, kg, acc);

  float* P = PART + (size_t)kc * BATCH * IN_DIM;
  const int n = n0 + row;
#pragma unroll
  for (int t = 0; t < 4; ++t)
#pragma unroll
    for (int r = 0; r < 8; ++r)
      P[(size_t)(m0 + t * 16 + kg * 8 + r) * IN_DIM + n] = acc[t][r];
}

// Fixed-order K-chunk reduction + bias + tanh (bitwise deterministic).
__global__ __launch_bounds__(256) void dec_reduce(const float* __restrict__ PART,
                                                  const float* __restrict__ Bv,
                                                  float* __restrict__ OUT) {
  const int n4 = (BATCH * IN_DIM) / 4;
  for (int idx = blockIdx.x * blockDim.x + threadIdx.x; idx < n4;
       idx += gridDim.x * blockDim.x) {
    f32x4 s = ((const f32x4*)PART)[idx];
#pragma unroll
    for (int c = 1; c < KCHUNKS; ++c)
      s += ((const f32x4*)(PART + (size_t)c * BATCH * IN_DIM))[idx];
    const f32x4 bb = *(const f32x4*)(Bv + (idx * 4) % IN_DIM);
    f32x4 r;
#pragma unroll
    for (int j = 0; j < 4; ++j) r[j] = tanhf(s[j] + bb[j]);
    ((f32x4*)OUT)[idx] = r;
  }
}

// ---------------------------------------------------------------------------
extern "C" void kernel_launch(void* const* d_in, const int* in_sizes, int n_in,
                              void* d_out, int out_size, void* d_ws, size_t ws_size,
                              hipStream_t stream) {
  const float* x     = (const float*)d_in[0];
  const float* W_enc = (const float*)d_in[1];
  const float* b_enc = (const float*)d_in[2];
  const float* W_dec = (const float*)d_in[3];
  const float* b_dec = (const float*)d_in[4];
  float* out = (float*)d_out;

  char* ws = (char*)d_ws;
  float*    pre  = (float*)ws;                                  // 41,943,040 B
  __bf16*   latb = (__bf16*)(ws + 41943040ull);                 // 20,971,520 B
  __bf16*   xb   = (__bf16*)(ws + 62914560ull);                 //  2,097,152 B
  float*    part = (float*)(ws + 65011712ull);                  // 33,554,432 B
  unsigned* hist = (unsigned*)(ws + 98566144ull);               //      8,192 B
  unsigned* sel  = hist + 2048;

  // x -> bf16 once.
  cvt_x<<<256, 256, 0, stream>>>(x, xb);

  // Encoder: 4 m-groups x 2560 n-tiles = 10240 waves (8 per block).
  enc_gemm<<<(4 * (LAT_DIM / 16)) / 8, 256, 0, stream>>>(xb, W_enc, b_enc, pre);

  // Global top-k threshold via 3-round radix select on monotone keys.
  topk_init<<<1, 256, 0, stream>>>(hist, sel);
  topk_hist<<<2048, 256, 0, stream>>>(pre, hist, sel, 21, 11);
  topk_scan<<<1, 1, 0, stream>>>(hist, sel, 21, 11, 0);
  topk_hist<<<2048, 256, 0, stream>>>(pre, hist, sel, 10, 11);
  topk_scan<<<1, 1, 0, stream>>>(hist, sel, 10, 11, 0);
  topk_hist<<<2048, 256, 0, stream>>>(pre, hist, sel, 0, 10);
  topk_scan<<<1, 1, 0, stream>>>(hist, sel, 0, 10, 1);
  topk_scatter<<<2048, 256, 0, stream>>>(pre, latb, sel);

  // Decoder: 4 m-groups x 8 k-chunks x 256 n-tiles = 8192 waves.
  dec_gemm<<<(4 * KCHUNKS * (IN_DIM / 16)) / 8, 256, 0, stream>>>(latb, W_dec, part);
  dec_reduce<<<1024, 256, 0, stream>>>(part, b_dec, out);
}
